// SMC_Transf_Cell_69947837383357
// MI455X (gfx1250) — matrix operand order repaired and checked by reading, not compiled, not run on hardware
//
#include <hip/hip_runtime.h>
#include <hip/hip_bf16.h>
#include <math.h>

typedef __attribute__((ext_vector_type(16))) __bf16 v16bf;
typedef __attribute__((ext_vector_type(8)))  float  v8f;

#define DEV static __device__ __forceinline__

constexpr int Bc = 32, Pc = 16, Sc = 512, Dc = 128, DFFc = 512, Vc = 32000;
constexpr int Nrows = Bc * Pc;          // 512 token rows
constexpr float SIGMAc = 0.05f;
constexpr float EPSc   = 1e-6f;

// ---------------- helpers ----------------
DEV unsigned short f2bf(float f) {
  unsigned x = __float_as_uint(f);
  unsigned r = ((x >> 16) & 1u) + 0x7FFFu;   // round-to-nearest-even
  return (unsigned short)((x + r) >> 16);
}
DEV unsigned pcg(unsigned v) {
  v = v * 747796405u + 2891336453u;
  unsigned w = ((v >> ((v >> 28) + 4u)) ^ v) * 277803737u;
  return (w >> 22) ^ w;
}
DEV float u01(unsigned v) { return (float)(pcg(v) >> 8) * (1.0f / 16777216.0f) + 1e-7f; }
DEV float gauss(unsigned s) {
  float u1 = u01(s * 2u + 1u), u2 = u01(s * 2u + 2u);
  return sqrtf(-2.0f * logf(u1)) * __cosf(6.28318530718f * u2);
}

// ---------------- f32 -> bf16 convert ----------------
__global__ void f32_to_bf16_kernel(const float* __restrict__ src,
                                   unsigned short* __restrict__ dst, int n) {
  for (int i = blockIdx.x * blockDim.x + threadIdx.x; i < n; i += gridDim.x * blockDim.x)
    dst[i] = f2bf(src[i]);
}

// ---------------- bf16 WMMA GEMM: C[M,N] = A[M,K]@B[K,N] (+bias,relu,noise) ----------
// grid = (M/16, N/128), block = 256 (8 waves; wave w owns n-tile w*16)
__global__ void wmma_gemm_bf16(const unsigned short* __restrict__ A,
                               const unsigned short* __restrict__ Bm,
                               const float* __restrict__ bias,
                               float* __restrict__ C,
                               unsigned short* __restrict__ Cbf,
                               int Kk, int Nn, int relu_flag,
                               float sigma, unsigned seed) {
  __shared__ unsigned As32[16][257];   // 16 rows x (K/2 dwords), padded: bank = (m+k2)%64
  const int tid   = threadIdx.x;
  const int lane  = tid & 31;
  const int wave  = tid >> 5;
  const int tileM = blockIdx.x * 16;
  const int nt0   = blockIdx.y * 128 + wave * 16;

  // cooperatively stage the 16 x K A-tile (bf16) in LDS
  const int groups = Kk >> 3;          // 8 bf16 (16B) per group
  for (int it = tid; it < 16 * groups; it += 256) {
    int row = it / groups, cg = it % groups;
    const uint4 a4 = *reinterpret_cast<const uint4*>(A + (size_t)(tileM + row) * Kk + cg * 8);
    As32[row][cg * 4 + 0] = a4.x; As32[row][cg * 4 + 1] = a4.y;
    As32[row][cg * 4 + 2] = a4.z; As32[row][cg * 4 + 3] = a4.w;
  }
  __syncthreads();

  const int m  = lane & 15;
  const int kh = (lane >> 4) * 8;      // ISA A-layout: lanes 16-31 hold K+8 / K+24 halves

  union Frag { v16bf v; unsigned u[8]; };
  v8f acc = {};

  const int nchunks = Kk >> 5;         // K steps of 32
  for (int c = 0; c < nchunks; ++c) {
    Frag af, bfr;
#pragma unroll
    for (int i = 0; i < 8; ++i) {      // A 16x32 bf16 layout (ISA 7.12.2)
      int koff = (i < 4) ? (kh + 2 * i) : (16 + kh + 2 * (i - 4));
      af.u[i] = As32[m][c * 16 + (koff >> 1)];
    }
    // B 32x16: lane&15 = N column, elements over K in the same order as A
    const int bn = nt0 + (lane & 15);
#pragma unroll
    for (int i = 0; i < 8; ++i) {
      int kb = c * 32 + ((i < 4) ? (kh + 2 * i) : (16 + kh + 2 * (i - 4)));
      bfr.u[i] = (unsigned)Bm[(size_t)kb * Nn + bn] | ((unsigned)Bm[(size_t)(kb + 1) * Nn + bn] << 16);
    }
    acc = __builtin_amdgcn_wmma_f32_16x16x32_bf16(false, af.v, false, bfr.v,
                                                  (short)0, acc, false, false);
  }

  // C layout (ISA 7.12.2): VGPR i -> row i (lanes 0-15) / row 8+i (lanes 16-31), col = lane&15
  const int rowOff = (lane >> 4) * 8;
  const int col    = nt0 + (lane & 15);
#pragma unroll
  for (int i = 0; i < 8; ++i) {
    int row = tileM + i + rowOff;
    float v = acc[i];
    if (bias)      v += bias[col];
    if (relu_flag) v = fmaxf(v, 0.f);
    if (sigma > 0.f) v += sigma * gauss(seed + (unsigned)(row * Nn + col));
    C[(size_t)row * Nn + col] = v;
    if (Cbf) Cbf[(size_t)row * Nn + col] = f2bf(v);
  }
}

// ---------------- layernorm over D=128 (optional residual) ----------------
__global__ void ln_kernel(const float* __restrict__ X, const float* __restrict__ Res,
                          const float* __restrict__ g, const float* __restrict__ bta,
                          float* __restrict__ Yf, unsigned short* __restrict__ Ybf) {
  __shared__ float sm[128];
  __shared__ float mu_s, rs_s;
  const int n = blockIdx.x, t = threadIdx.x;
  float v = X[n * Dc + t];
  if (Res) v += Res[n * Dc + t];
  sm[t] = v; __syncthreads();
  for (int o = 64; o > 0; o >>= 1) { if (t < o) sm[t] += sm[t + o]; __syncthreads(); }
  if (t == 0) mu_s = sm[0] * (1.0f / Dc);
  __syncthreads();
  float d = v - mu_s;
  sm[t] = d * d; __syncthreads();
  for (int o = 64; o > 0; o >>= 1) { if (t < o) sm[t] += sm[t + o]; __syncthreads(); }
  if (t == 0) rs_s = rsqrtf(sm[0] * (1.0f / Dc) + EPSc);
  __syncthreads();
  float y = d * rs_s * g[t] + bta[t];
  Yf[n * Dc + t] = y;
  if (Ybf) Ybf[n * Dc + t] = f2bf(y);
}

// ---------------- attention: per (b,p) GEMV + softmax + attn@V ----------------
__global__ void attn_kernel(const float* __restrict__ q,
                            const float* __restrict__ Kin,
                            const float* __restrict__ Vin,
                            const float* __restrict__ knew,
                            const float* __restrict__ vnew,
                            const int* __restrict__ tptr,
                            float* __restrict__ attn_out,    // [N,S]
                            float* __restrict__ av_f,        // [N,D]
                            unsigned short* __restrict__ av_b) {
  __shared__ float qs[Dc];
  __shared__ float lg[Sc];
  __shared__ float red[128];
  __shared__ float mx_s, sum_s;
  const int n = blockIdx.x, t = threadIdx.x;
  const int T = *tptr;
  qs[t] = q[n * Dc + t];
  __syncthreads();
  for (int s = t; s <= T; s += 128) {
    const float* Kr = (s == T) ? (knew + (size_t)n * Dc)
                               : (Kin + ((size_t)n * Sc + s) * Dc);
    float acc = 0.f;
#pragma unroll 8
    for (int d = 0; d < Dc; d += 4)
      acc += qs[d] * Kr[d] + qs[d + 1] * Kr[d + 1] + qs[d + 2] * Kr[d + 2] + qs[d + 3] * Kr[d + 3];
    lg[s] = acc * 0.0883883476f;   // 1/sqrt(128)
  }
  __syncthreads();
  float lmx = -3.0e38f;
  for (int s = t; s <= T; s += 128) lmx = fmaxf(lmx, lg[s]);
  red[t] = lmx; __syncthreads();
  for (int o = 64; o > 0; o >>= 1) { if (t < o) red[t] = fmaxf(red[t], red[t + o]); __syncthreads(); }
  if (t == 0) mx_s = red[0];
  __syncthreads();
  float lsum = 0.f;
  for (int s = t; s <= T; s += 128) lsum += expf(lg[s] - mx_s);
  red[t] = lsum; __syncthreads();
  for (int o = 64; o > 0; o >>= 1) { if (t < o) red[t] += red[t + o]; __syncthreads(); }
  if (t == 0) sum_s = red[0];
  __syncthreads();
  float inv = 1.0f / sum_s;
  for (int s = t; s < Sc; s += 128) {
    float a = (s <= T) ? expf(lg[s] - mx_s) * inv : 0.f;
    attn_out[(size_t)n * Sc + s] = a;
    lg[s] = a;
  }
  __syncthreads();
  float acc = 0.f;
  for (int s = 0; s <= T; ++s) {
    float vv = (s == T) ? vnew[(size_t)n * Dc + t] : Vin[((size_t)n * Sc + s) * Dc + t];
    acc += lg[s] * vv;
  }
  av_f[n * Dc + t] = acc;
  av_b[n * Dc + t] = f2bf(acc);
}

// ---------------- vocab softmax -> w[n] = probs[y[n]] ----------------
__global__ void vocab_w_kernel(const float* __restrict__ preds, const int* __restrict__ y,
                               float* __restrict__ wout) {
  __shared__ float red[256];
  __shared__ float mx_s;
  const int n = blockIdx.x, t = threadIdx.x;
  const float* p = preds + (size_t)n * Vc;
  float lmx = -3.0e38f;
  for (int j = t; j < Vc; j += 256) lmx = fmaxf(lmx, p[j]);
  red[t] = lmx; __syncthreads();
  for (int o = 128; o > 0; o >>= 1) { if (t < o) red[t] = fmaxf(red[t], red[t + o]); __syncthreads(); }
  if (t == 0) mx_s = red[0];
  __syncthreads();
  float ls = 0.f;
  for (int j = t; j < Vc; j += 256) ls += expf(p[j] - mx_s);
  red[t] = ls; __syncthreads();
  for (int o = 128; o > 0; o >>= 1) { if (t < o) red[t] += red[t + o]; __syncthreads(); }
  if (t == 0) {
    int yv = y[n]; yv = (yv < 0) ? 0 : ((yv >= Vc) ? Vc - 1 : yv);
    wout[n] = expf(p[yv] - mx_s) / red[0];
  }
}

// ---------------- categorical resampling (w used as logits) ----------------
__global__ void sample_kernel(const float* __restrict__ w, int* __restrict__ idx) {
  const int b = blockIdx.x;
  if (threadIdx.x != 0) return;
  float mx = -3.0e38f;
  for (int p = 0; p < Pc; ++p) mx = fmaxf(mx, w[b * Pc + p]);
  float pr[Pc]; float s = 0.f;
  for (int p = 0; p < Pc; ++p) { pr[p] = expf(w[b * Pc + p] - mx); s += pr[p]; }
  for (int pn = 0; pn < Pc; ++pn) {
    float u = u01(0x9e3779b9u + (unsigned)(b * Pc + pn)) * s;
    float cum = 0.f; int sel = Pc - 1;
    for (int p = 0; p < Pc; ++p) { cum += pr[p]; if (u <= cum) { sel = p; break; } }
    idx[b * Pc + pn] = sel;
  }
}

// -------- fused particle-gather + dynamic_update_slice (row T <- newrow) --------
__global__ void gather_kernel(const float* __restrict__ src,     // [N,S,D] input cache
                              const float* __restrict__ newrow,  // [N,D]
                              const int* __restrict__ idx,
                              const int* __restrict__ tptr,
                              float* __restrict__ dst) {
  const int T = *tptr;
  const size_t total = (size_t)Nrows * Sc * (Dc / 4);
  const float4* s4 = reinterpret_cast<const float4*>(src);
  const float4* n4 = reinterpret_cast<const float4*>(newrow);
  float4* d4 = reinterpret_cast<float4*>(dst);
  for (size_t e = blockIdx.x * (size_t)blockDim.x + threadIdx.x; e < total;
       e += (size_t)gridDim.x * blockDim.x) {
    int c4 = (int)(e & 31);            // D/4 = 32
    size_t t1 = e >> 5;
    int s  = (int)(t1 & 511);          // S = 512
    int np = (int)(t1 >> 9);
    int b  = np >> 4;                  // P = 16
    int nsrc = (b << 4) | idx[np];
    float4 v = (s == T) ? n4[(size_t)nsrc * 32 + c4]
                        : s4[((size_t)nsrc * Sc + s) * 32 + c4];
    d4[e] = v;
  }
}

// ---------------- host-side launch ----------------
extern "C" void kernel_launch(void* const* d_in, const int* in_sizes, int n_in,
                              void* d_out, int out_size, void* d_ws, size_t ws_size,
                              hipStream_t stream) {
  (void)in_sizes; (void)n_in; (void)out_size; (void)ws_size;
  const float* x    = (const float*)d_in[0];
  const int*   y    = (const int*)d_in[1];
  const int*   tp   = (const int*)d_in[2];
  const float* Kin  = (const float*)d_in[3];
  const float* Vin  = (const float*)d_in[4];
  const float* Rin  = (const float*)d_in[5];
  const float* ln1g = (const float*)d_in[6];
  const float* ln1b = (const float*)d_in[7];
  const float* ln2g = (const float*)d_in[8];
  const float* ln2b = (const float*)d_in[9];
  const float* Wq = (const float*)d_in[10]; const float* bq = (const float*)d_in[11];
  const float* Wk = (const float*)d_in[12]; const float* bk = (const float*)d_in[13];
  const float* Wv = (const float*)d_in[14]; const float* bv = (const float*)d_in[15];
  const float* Wz = (const float*)d_in[16]; const float* bz = (const float*)d_in[17];
  const float* W1 = (const float*)d_in[18]; const float* b1 = (const float*)d_in[19];
  const float* W2 = (const float*)d_in[20]; const float* b2 = (const float*)d_in[21];
  const float* Wout = (const float*)d_in[22];

  // ---- workspace layout (byte offsets) ----
  char* wsb = (char*)d_ws;
  float* xqf  = (float*)(wsb + 0);                 // 512x128 f32
  unsigned short* xqb = (unsigned short*)(wsb + 262144);
  float* qf   = (float*)(wsb + 393216);
  float* kf   = (float*)(wsb + 655360);
  float* vf   = (float*)(wsb + 917504);
  float* avf  = (float*)(wsb + 1179648);
  unsigned short* avb = (unsigned short*)(wsb + 1441792);
  float* zpf  = (float*)(wsb + 1572864);
  float* outf = (float*)(wsb + 1835008);
  unsigned short* outb = (unsigned short*)(wsb + 2097152);
  float* hf   = (float*)(wsb + 2228224);           // 512x512 f32
  unsigned short* hb = (unsigned short*)(wsb + 3276800);
  float* r0f  = (float*)(wsb + 3801088);
  unsigned short* rb = (unsigned short*)(wsb + 4063232);
  unsigned short* wqb2 = (unsigned short*)(wsb + 4194304);
  unsigned short* wkb2 = (unsigned short*)(wsb + 4227072);
  unsigned short* wvb2 = (unsigned short*)(wsb + 4259840);
  unsigned short* wzb2 = (unsigned short*)(wsb + 4292608);
  unsigned short* w1b2 = (unsigned short*)(wsb + 4325376);
  unsigned short* w2b2 = (unsigned short*)(wsb + 4456448);
  unsigned short* wob2 = (unsigned short*)(wsb + 4587520);   // 128x32000 bf16
  int*   idxbuf = (int*)(wsb + 12779520);
  float* preds  = (float*)(wsb + 12781568);                  // 512x32000 f32

  // ---- output layout (float offsets, return order: r, attn, w, K, V, R) ----
  float* ro    = (float*)d_out;          // 512*128
  float* ao    = ro + 65536;             // 512*512
  float* wo    = ro + 327680;            // 512
  float* Ko    = ro + 328192;            // 512*512*128
  float* Vo    = ro + 33882624;
  float* Ro    = ro + 67437056;

  // 1) weights -> bf16
  f32_to_bf16_kernel<<<64,   256, 0, stream>>>(Wq, wqb2, Dc * Dc);
  f32_to_bf16_kernel<<<64,   256, 0, stream>>>(Wk, wkb2, Dc * Dc);
  f32_to_bf16_kernel<<<64,   256, 0, stream>>>(Wv, wvb2, Dc * Dc);
  f32_to_bf16_kernel<<<64,   256, 0, stream>>>(Wz, wzb2, Dc * Dc);
  f32_to_bf16_kernel<<<256,  256, 0, stream>>>(W1, w1b2, Dc * DFFc);
  f32_to_bf16_kernel<<<256,  256, 0, stream>>>(W2, w2b2, DFFc * Dc);
  f32_to_bf16_kernel<<<4096, 256, 0, stream>>>(Wout, wob2, Dc * Vc);

  // 2) xq = LN1(x)
  ln_kernel<<<Nrows, 128, 0, stream>>>(x, nullptr, ln1g, ln1b, xqf, xqb);

  // 3) q/k/v projections (WMMA) + noise
  wmma_gemm_bf16<<<dim3(Nrows / 16, 1), 256, 0, stream>>>(xqb, wqb2, bq, qf, nullptr, Dc, Dc, 0, SIGMAc, 0x11110000u);
  wmma_gemm_bf16<<<dim3(Nrows / 16, 1), 256, 0, stream>>>(xqb, wkb2, bk, kf, nullptr, Dc, Dc, 0, SIGMAc, 0x22220000u);
  wmma_gemm_bf16<<<dim3(Nrows / 16, 1), 256, 0, stream>>>(xqb, wvb2, bv, vf, nullptr, Dc, Dc, 0, SIGMAc, 0x33330000u);

  // 4) attention (causal, row T comes from new k/v)
  attn_kernel<<<Nrows, 128, 0, stream>>>(qf, Kin, Vin, kf, vf, tp, ao, avf, avb);

  // 5) z = (attn@V)@Wz + bz + noise; out = LN1(z + xq)
  wmma_gemm_bf16<<<dim3(Nrows / 16, 1), 256, 0, stream>>>(avb, wzb2, bz, zpf, nullptr, Dc, Dc, 0, SIGMAc, 0x44440000u);
  ln_kernel<<<Nrows, 128, 0, stream>>>(zpf, xqf, ln1g, ln1b, outf, outb);

  // 6) FFN: h = relu(out@W1+b1); r0 = h@W2+b2; r = LN2(r0 + out) -> d_out r region
  wmma_gemm_bf16<<<dim3(Nrows / 16, DFFc / 128), 256, 0, stream>>>(outb, w1b2, b1, hf, hb, Dc, DFFc, 1, 0.f, 0u);
  wmma_gemm_bf16<<<dim3(Nrows / 16, 1), 256, 0, stream>>>(hb, w2b2, b2, r0f, nullptr, DFFc, Dc, 0, 0.f, 0u);
  ln_kernel<<<Nrows, 128, 0, stream>>>(r0f, outf, ln2g, ln2b, ro, rb);

  // 7) vocab head: preds = r @ Wout (512 x 32000, WMMA)
  wmma_gemm_bf16<<<dim3(Nrows / 16, Vc / 128), 256, 0, stream>>>(rb, wob2, nullptr, preds, nullptr, Dc, Vc, 0, 0.f, 0u);

  // 8) w = softmax(preds)[y]
  vocab_w_kernel<<<Nrows, 256, 0, stream>>>(preds, y, wo);

  // 9) resample particle indices
  sample_kernel<<<Bc, 32, 0, stream>>>(wo, idxbuf);

  // 10) fused gather + timestep-row update for K, V, R
  gather_kernel<<<4096, 256, 0, stream>>>(Kin, kf, idxbuf, tp, Ko);
  gather_kernel<<<4096, 256, 0, stream>>>(Vin, vf, idxbuf, tp, Vo);
  gather_kernel<<<4096, 256, 0, stream>>>(Rin, ro, idxbuf, tp, Ro);
}
